// Encoder_79259326480726
// MI455X (gfx1250) — matrix operand order repaired and checked by reading, not compiled, not run on hardware
//
#include <hip/hip_runtime.h>
#include <math.h>
#include <stdint.h>

// ---------------- types & helpers ----------------
typedef __bf16 bf16_t;
typedef __attribute__((ext_vector_type(16))) __bf16 v16bf;
typedef __attribute__((ext_vector_type(8)))  float  v8f;

union FragA { uint4 u[2]; v16bf v; };

static __device__ __forceinline__ bf16_t f2bf(float f) {
    union { float f; unsigned u; } a; a.f = f;
    unsigned r = (a.u + 0x7fffu + ((a.u >> 16) & 1u)) >> 16;
    unsigned short s = (unsigned short)r;
    return __builtin_bit_cast(bf16_t, s);
}
static __device__ __forceinline__ float bf2f(bf16_t b) {
    unsigned short s = __builtin_bit_cast(unsigned short, b);
    unsigned u = ((unsigned)s) << 16;
    union { unsigned u; float f; } a; a.u = u;
    return a.f;
}

// CDNA5 async global->LDS copy (16B per lane), tracked by ASYNCcnt.
static __device__ __forceinline__ void async_b128(const void* gsrc, void* ldst) {
    unsigned long long ga = (unsigned long long)(uintptr_t)gsrc;
    unsigned loff = (unsigned)(uintptr_t)ldst;   // low 32 bits = LDS offset
    asm volatile("global_load_async_to_lds_b128 %0, %1, off"
                 :: "v"(loff), "v"(ga) : "memory");
}
static __device__ __forceinline__ void wait_async_le4() {
    asm volatile("s_wait_asynccnt 0x4" ::: "memory");
}
static __device__ __forceinline__ void wait_async_le2() {
    asm volatile("s_wait_asynccnt 0x2" ::: "memory");
}
static __device__ __forceinline__ void wait_async_0() {
    asm volatile("s_wait_asynccnt 0x0" ::: "memory");
}

#define EPSF 1e-5f

// Shapes (hard-coded to reference problem)
#define BATCH 16
#define HH_ 48
#define WW_ 48
#define NSEQ 2304           // 48*48
#define CDIM 384
#define C2DIM 768
#define C3DIM 1152
#define MROWS (BATCH*NSEQ)  // 36864
#define HEADS 8
#define DHEAD 48

// ---------------- weight prep ----------------
// dst[k*O + o] = src[o*K + k], fp32 -> bf16
__global__ void wtrans_kernel(const float* __restrict__ src, bf16_t* __restrict__ dst,
                              int O, int K) {
    int i = blockIdx.x * 256 + threadIdx.x;
    if (i < O * K) {
        int o = i / K; int k = i - o * K;
        dst[(size_t)k * O + o] = f2bf(src[i]);
    }
}

// conv weight (O,C,3,3) -> dst[(tap*C + c)*O + o]
__global__ void wconv_kernel(const float* __restrict__ src, bf16_t* __restrict__ dst) {
    int i = blockIdx.x * 256 + threadIdx.x;
    if (i < CDIM * CDIM * 9) {
        int o = i / (CDIM * 9);
        int rem = i - o * CDIM * 9;
        int c = rem / 9;
        int t = rem - c * 9;
        dst[((size_t)t * CDIM + c) * CDIM + o] = f2bf(src[i]);
    }
}

// ---------------- LayerNorm (fp32 in -> bf16 out) ----------------
__global__ __launch_bounds__(128) void ln_bf16_kernel(const float* __restrict__ x,
                                                      const float* __restrict__ g,
                                                      const float* __restrict__ b,
                                                      bf16_t* __restrict__ out) {
    int row = blockIdx.x;
    int t = threadIdx.x;
    const float* xr = x + (size_t)row * CDIM;
    float s = 0.f, s2 = 0.f;
    for (int i = t; i < CDIM; i += 128) { float v = xr[i]; s += v; s2 += v * v; }
    __shared__ float rs[128], rs2[128];
    rs[t] = s; rs2[t] = s2; __syncthreads();
    for (int off = 64; off; off >>= 1) {
        if (t < off) { rs[t] += rs[t + off]; rs2[t] += rs2[t + off]; }
        __syncthreads();
    }
    float m = rs[0] * (1.f / CDIM);
    float var = rs2[0] * (1.f / CDIM) - m * m;
    float inv = rsqrtf(var + EPSF);
    bf16_t* orow = out + (size_t)row * CDIM;
    for (int i = t; i < CDIM; i += 128)
        orow[i] = f2bf((xr[i] - m) * inv * g[i] + b[i]);
}

// ---------------- q/k L2 norms over N ----------------
__global__ __launch_bounds__(128) void qknorm_kernel(const bf16_t* __restrict__ qkv,
                                                     float* __restrict__ invq,
                                                     float* __restrict__ invk) {
    int id = blockIdx.x;             // 2 * B * heads * d
    int dd = id % DHEAD; int rest = id / DHEAD;
    int head = rest % HEADS; rest /= HEADS;
    int b = rest % BATCH; int s = rest / BATCH;   // 0=q, 1=k
    const bf16_t* base = qkv + (size_t)b * NSEQ * C3DIM + s * CDIM + head * DHEAD + dd;
    int t = threadIdx.x;
    float ss = 0.f;
    for (int n = t; n < NSEQ; n += 128) {
        float v = bf2f(base[(size_t)n * C3DIM]);
        ss += v * v;
    }
    __shared__ float red[128];
    red[t] = ss; __syncthreads();
    for (int off = 64; off; off >>= 1) { if (t < off) red[t] += red[t + off]; __syncthreads(); }
    if (t == 0) {
        float inv = 1.f / fmaxf(sqrtf(red[0]), 1e-12f);
        (s ? invk : invq)[(b * HEADS + head) * DHEAD + dd] = inv;
    }
}

// ---------------- channel attention per (b, head) ----------------
__global__ __launch_bounds__(128) void attn_kernel(const bf16_t* __restrict__ qkv,
                                                   const float* __restrict__ invq,
                                                   const float* __restrict__ invk,
                                                   const float* __restrict__ temp,
                                                   bf16_t* __restrict__ out) {
    int blk = blockIdx.x;
    int head = blk % HEADS;
    int b = blk / HEADS;
    int tid = threadIdx.x;
    int lane = tid & 31, w = tid >> 5;
    int lidx = lane & 15, lh = lane >> 4;

    __shared__ bf16_t qs[DHEAD * 40];
    __shared__ bf16_t ks[DHEAD * 40];
    __shared__ float  af[DHEAD * 49];
    __shared__ bf16_t ab[DHEAD * 64];

    const bf16_t* qbase = qkv + (size_t)b * NSEQ * C3DIM + head * DHEAD;
    const bf16_t* kbase = qbase + CDIM;
    const bf16_t* vbase = qbase + 2 * CDIM;
    const float* iq = invq + (b * HEADS + head) * DHEAD;
    const float* ik = invk + (b * HEADS + head) * DHEAD;

    // Stage 1: attn_logits[48x48] = q_hat[48 x N] * k_hat^T
    v8f acc[3] = {};
    for (int kt = 0; kt < NSEQ / 32; ++kt) {
        __syncthreads();
        for (int idx = tid; idx < DHEAD * 32; idx += 128) {
            int dd = idx % DHEAD; int n = idx / DHEAD;
            size_t go = (size_t)(kt * 32 + n) * C3DIM + dd;
            qs[dd * 40 + n] = f2bf(bf2f(qbase[go]) * iq[dd]);
            ks[dd * 40 + n] = f2bf(bf2f(kbase[go]) * ik[dd]);
        }
        __syncthreads();
        if (w < 3) {
            v16bf afr;
            {
                int row = w * 16 + lidx;
                const bf16_t* p = &qs[row * 40 + lh * 8];
                for (int h = 0; h < 8; h++) { afr[h] = p[h]; afr[8 + h] = p[16 + h]; }
            }
            for (int j = 0; j < 3; j++) {
                v16bf bfr;
                int col = j * 16 + lidx;
                const bf16_t* p = &ks[col * 40 + lh * 8];
                for (int h = 0; h < 8; h++) { bfr[h] = p[h]; bfr[8 + h] = p[16 + h]; }
                acc[j] = __builtin_amdgcn_wmma_f32_16x16x32_bf16(
                    false, afr, false, bfr, (short)0, acc[j], false, false);
            }
        }
    }
    __syncthreads();
    float tsc = temp[head];
    if (w < 3) {
        for (int j = 0; j < 3; j++)
            for (int r = 0; r < 8; r++) {
                int m = w * 16 + r + lh * 8;
                int cc = j * 16 + lidx;
                af[m * 49 + cc] = acc[j][r] * tsc;
            }
    }
    __syncthreads();
    // softmax per row, write bf16 attn padded to K=64
    if (tid < DHEAD) {
        float mx = -1e30f;
        for (int e = 0; e < DHEAD; e++) mx = fmaxf(mx, af[tid * 49 + e]);
        float sm = 0.f;
        for (int e = 0; e < DHEAD; e++) { float ex = __expf(af[tid * 49 + e] - mx); af[tid * 49 + e] = ex; sm += ex; }
        float inv = 1.f / sm;
        for (int e = 0; e < DHEAD; e++) ab[tid * 64 + e] = f2bf(af[tid * 49 + e] * inv);
        for (int e = DHEAD; e < 64; e++) ab[tid * 64 + e] = f2bf(0.f);
    }
    __syncthreads();
    // Stage 2: out[48 x N] = attn[48x48(pad64)] * v[48 x N]
    for (int nt = 0; nt < NSEQ / 64; ++nt) {
        int n = nt * 64 + w * 16 + lidx;
        v8f acc2[3] = {};
        const bf16_t* vp = vbase + (size_t)n * C3DIM;
        for (int kk = 0; kk < 2; ++kk) {
            v16bf bfr;
            for (int h = 0; h < 16; h++) {
                int e = kk * 32 + lh * 16 + h;
                bfr[h] = (e < DHEAD) ? vp[e] : f2bf(0.f);
            }
            for (int i = 0; i < 3; i++) {
                v16bf afr;
                int row = i * 16 + lidx;
                for (int h = 0; h < 8; h++) {
                    afr[h]     = ab[row * 64 + kk * 32 + lh * 16 + h];
                    afr[8 + h] = ab[row * 64 + kk * 32 + lh * 16 + 8 + h];
                }
                acc2[i] = __builtin_amdgcn_wmma_f32_16x16x32_bf16(
                    false, afr, false, bfr, (short)0, acc2[i], false, false);
            }
        }
        bf16_t* op = out + ((size_t)b * NSEQ + n) * CDIM + head * DHEAD;
        for (int i = 0; i < 3; i++)
            for (int r = 0; r < 8; r++)
                op[i * 16 + r + lh * 8] = f2bf(acc2[i][r]);
    }
}

// ---------------- generic WMMA GEMM with fused epilogues ----------------
// C[M,N] = A[M,K](bf16) * B[K,N](bf16).  CONV: A is NHWC, K = 9*CDIM implicit im2col.
// Double-buffered LDS tiles; B slab (and A slab in non-conv mode) is filled with
// global_load_async_to_lds_b128 overlapped with the previous tile's WMMA work.
struct EpiArgs {
    const float* bn_g; const float* bn_b; const float* bn_m; const float* bn_v;
    const float* gamma; const float* resid;   // fp32 [M,N]
    float* out_f32; bf16_t* out_bf16;
};
// EPI: 0 = store bf16; 1 = resid + gamma*acc -> f32; 2 = BN+GELU -> bf16; 3 = BN then resid+gamma -> f32
template <int EPI, bool CONV>
__global__ __launch_bounds__(256) void gemm_kernel(const bf16_t* __restrict__ A,
                                                   const bf16_t* __restrict__ Bm,
                                                   int M, int N, int K, EpiArgs ep) {
    __shared__ __align__(16) bf16_t As[2][128 * 48];
    __shared__ __align__(16) bf16_t Bs[2][32 * 144];
    int tid = threadIdx.x;
    int lane = tid & 31, wid = tid >> 5;
    int lidx = lane & 15, lh = lane >> 4;
    int wm = wid & 1, wn = wid >> 1;                // 2x4 wave grid, wave tile 64x32
    int n0 = blockIdx.x * 128, m0 = blockIdx.y * 128;
    v8f acc[4][2] = {};

    int arow = tid >> 1, ahalf = (tid & 1) * 16;    // A stage: 2 threads/row, 16 bf16 each
    int brow = tid >> 3, bcol = (tid & 7) * 16;     // B stage: 8 threads/row
    int KT = K / 32;

    auto stage = [&](int kt, int buf) {
        int k0 = kt * 32;
        // B slab: always valid -> async copy (2 x b128 per lane)
        {
            const bf16_t* src = Bm + ((size_t)(k0 + brow) * N + n0 + bcol);
            bf16_t* dst = &Bs[buf][brow * 144 + bcol];
            async_b128(src, dst);
            async_b128(src + 8, dst + 8);
        }
        // A slab
        if (CONV) {
            int tap = k0 / CDIM; int c0 = k0 - tap * CDIM;
            int ky = tap / 3 - 1, kx = tap % 3 - 1;
            int gr = m0 + arow;
            int bb = gr / NSEQ; int hw = gr - bb * NSEQ;
            int hh = hw / WW_ + ky, ww = hw % WW_ + kx;
            bool valid = (hh >= 0 && hh < HH_ && ww >= 0 && ww < WW_);
            const bf16_t* src = A + ((size_t)(bb * NSEQ + hh * WW_ + ww) * CDIM + c0 + ahalf);
            uint4* d = (uint4*)&As[buf][arow * 48 + ahalf];
            if (valid) { const uint4* s = (const uint4*)src; d[0] = s[0]; d[1] = s[1]; }
            else       { uint4 z = make_uint4(0u, 0u, 0u, 0u); d[0] = z; d[1] = z; }
        } else {
            const bf16_t* src = A + ((size_t)(m0 + arow) * K + k0 + ahalf);
            bf16_t* dst = &As[buf][arow * 48 + ahalf];
            async_b128(src, dst);
            async_b128(src + 8, dst + 8);
        }
    };

    stage(0, 0);
    for (int kt = 0; kt < KT; ++kt) {
        int cur = kt & 1;
        if (kt + 1 < KT) {
            stage(kt + 1, cur ^ 1);
            if (CONV) wait_async_le2(); else wait_async_le4();
        } else {
            wait_async_0();
        }
        __syncthreads();

        v16bf afr[4], bfr[2];
        for (int i = 0; i < 4; i++) {
            const bf16_t* p = &As[cur][(wm * 64 + i * 16 + lidx) * 48 + lh * 8];
            FragA f; f.u[0] = *(const uint4*)p; f.u[1] = *(const uint4*)(p + 16);
            afr[i] = f.v;
        }
        for (int j = 0; j < 2; j++) {
            int col = wn * 32 + j * 16 + lidx;
            v16bf bv;
            for (int h = 0; h < 16; h++) bv[h] = Bs[cur][(h < 8 ? lh * 8 + h : 8 + lh * 8 + h) * 144 + col];
            bfr[j] = bv;
        }
        for (int i = 0; i < 4; i++)
            for (int j = 0; j < 2; j++)
                acc[i][j] = __builtin_amdgcn_wmma_f32_16x16x32_bf16(
                    false, afr[i], false, bfr[j], (short)0, acc[i][j], false, false);
        __syncthreads();   // protect buffer reuse before next prefetch
    }

    // epilogue
    for (int i = 0; i < 4; i++)
        for (int j = 0; j < 2; j++) {
            int gn = n0 + wn * 32 + j * 16 + lidx;
            for (int r = 0; r < 8; r++) {
                int gm = m0 + wm * 64 + i * 16 + r + lh * 8;
                size_t idx = (size_t)gm * N + gn;
                float v = acc[i][j][r];
                if (EPI == 0) {
                    ep.out_bf16[idx] = f2bf(v);
                } else if (EPI == 1) {
                    ep.out_f32[idx] = ep.resid[idx] + ep.gamma[gn] * v;
                } else if (EPI == 2) {
                    float t = (v - ep.bn_m[gn]) * rsqrtf(ep.bn_v[gn] + EPSF) * ep.bn_g[gn] + ep.bn_b[gn];
                    float g = 0.5f * t * (1.f + erff(t * 0.70710678118654752f));
                    ep.out_bf16[idx] = f2bf(g);
                } else {
                    float t = (v - ep.bn_m[gn]) * rsqrtf(ep.bn_v[gn] + EPSF) * ep.bn_g[gn] + ep.bn_b[gn];
                    ep.out_f32[idx] = ep.resid[idx] + ep.gamma[gn] * t;
                }
            }
        }
}

// ---------------- tail: write H, W scalars ----------------
__global__ void tail_kernel(float* out, const int* Hp, const int* Wp, int out_size) {
    if (threadIdx.x == 0 && blockIdx.x == 0 && out_size >= MROWS * CDIM + 2) {
        out[MROWS * CDIM + 0] = (float)Hp[0];
        out[MROWS * CDIM + 1] = (float)Wp[0];
    }
}

// ---------------- launcher ----------------
extern "C" void kernel_launch(void* const* d_in, const int* in_sizes, int n_in,
                              void* d_out, int out_size, void* d_ws, size_t ws_size,
                              hipStream_t stream) {
    (void)in_sizes; (void)n_in; (void)ws_size;
    const float* x      = (const float*)d_in[0];
    const float* ln1_g  = (const float*)d_in[1];
    const float* ln1_b  = (const float*)d_in[2];
    const float* w_qkv  = (const float*)d_in[3];
    const float* temp   = (const float*)d_in[4];
    const float* w_proj = (const float*)d_in[5];
    const float* gamma1 = (const float*)d_in[6];
    const float* ln2_g  = (const float*)d_in[7];
    const float* ln2_b  = (const float*)d_in[8];
    const float* mlp_w1 = (const float*)d_in[9];
    const float* bn1_g  = (const float*)d_in[10];
    const float* bn1_b  = (const float*)d_in[11];
    const float* bn1_m  = (const float*)d_in[12];
    const float* bn1_v  = (const float*)d_in[13];
    const float* mlp_w2 = (const float*)d_in[14];
    const float* bn2_g  = (const float*)d_in[15];
    const float* bn2_b  = (const float*)d_in[16];
    const float* bn2_m  = (const float*)d_in[17];
    const float* bn2_v  = (const float*)d_in[18];
    const float* gamma2 = (const float*)d_in[19];
    const float* ln3_g  = (const float*)d_in[20];
    const float* ln3_b  = (const float*)d_in[21];
    const float* pconv1 = (const float*)d_in[22];
    const float* pbn_g  = (const float*)d_in[23];
    const float* pbn_b  = (const float*)d_in[24];
    const float* pbn_m  = (const float*)d_in[25];
    const float* pbn_v  = (const float*)d_in[26];
    const float* pconv2 = (const float*)d_in[27];
    const float* gamma3 = (const float*)d_in[28];
    const int*   Hp     = (const int*)d_in[29];
    const int*   Wp     = (const int*)d_in[30];

    char* wsb = (char*)d_ws;
    size_t off = 0;
    auto alloc = [&](size_t bytes) -> char* {
        char* p = wsb + off;
        off += (bytes + 255) & ~(size_t)255;
        return p;
    };
    bf16_t* wq_t  = (bf16_t*)alloc((size_t)CDIM * C3DIM * 2);
    bf16_t* wp_t  = (bf16_t*)alloc((size_t)CDIM * CDIM * 2);
    bf16_t* w1_t  = (bf16_t*)alloc((size_t)CDIM * C2DIM * 2);
    bf16_t* w2_t  = (bf16_t*)alloc((size_t)C2DIM * CDIM * 2);
    bf16_t* wc1_t = (bf16_t*)alloc((size_t)9 * CDIM * CDIM * 2);
    bf16_t* wc2_t = (bf16_t*)alloc((size_t)9 * CDIM * CDIM * 2);
    bf16_t* ln_bf = (bf16_t*)alloc((size_t)MROWS * CDIM * 2);
    bf16_t* qkvbf = (bf16_t*)alloc((size_t)MROWS * C3DIM * 2);
    float*  invq  = (float*)alloc((size_t)BATCH * HEADS * DHEAD * 4);
    float*  invk  = (float*)alloc((size_t)BATCH * HEADS * DHEAD * 4);
    bf16_t* at_bf = (bf16_t*)alloc((size_t)MROWS * CDIM * 2);
    float*  x1    = (float*)alloc((size_t)MROWS * CDIM * 4);
    bf16_t* y1    = (bf16_t*)alloc((size_t)MROWS * C2DIM * 2);
    float*  x2    = (float*)alloc((size_t)MROWS * CDIM * 4);
    bf16_t* p1    = (bf16_t*)alloc((size_t)MROWS * CDIM * 2);

    // --- weight prep ---
    wtrans_kernel<<<(C3DIM * CDIM + 255) / 256, 256, 0, stream>>>(w_qkv, wq_t, C3DIM, CDIM);
    wtrans_kernel<<<(CDIM * CDIM + 255) / 256, 256, 0, stream>>>(w_proj, wp_t, CDIM, CDIM);
    wtrans_kernel<<<(C2DIM * CDIM + 255) / 256, 256, 0, stream>>>(mlp_w1, w1_t, C2DIM, CDIM);
    wtrans_kernel<<<(CDIM * C2DIM + 255) / 256, 256, 0, stream>>>(mlp_w2, w2_t, CDIM, C2DIM);
    wconv_kernel<<<(CDIM * CDIM * 9 + 255) / 256, 256, 0, stream>>>(pconv1, wc1_t);
    wconv_kernel<<<(CDIM * CDIM * 9 + 255) / 256, 256, 0, stream>>>(pconv2, wc2_t);

    EpiArgs ep{};

    // --- attention block ---
    ln_bf16_kernel<<<MROWS, 128, 0, stream>>>(x, ln1_g, ln1_b, ln_bf);
    ep = EpiArgs{}; ep.out_bf16 = qkvbf;
    gemm_kernel<0, false><<<dim3(C3DIM / 128, MROWS / 128), 256, 0, stream>>>(
        ln_bf, wq_t, MROWS, C3DIM, CDIM, ep);
    qknorm_kernel<<<2 * BATCH * HEADS * DHEAD, 128, 0, stream>>>(qkvbf, invq, invk);
    attn_kernel<<<BATCH * HEADS, 128, 0, stream>>>(qkvbf, invq, invk, temp, at_bf);
    ep = EpiArgs{}; ep.resid = x; ep.gamma = gamma1; ep.out_f32 = x1;
    gemm_kernel<1, false><<<dim3(CDIM / 128, MROWS / 128), 256, 0, stream>>>(
        at_bf, wp_t, MROWS, CDIM, CDIM, ep);

    // --- ConvMlp block (1x1 convs + BN) ---
    ln_bf16_kernel<<<MROWS, 128, 0, stream>>>(x1, ln2_g, ln2_b, ln_bf);
    ep = EpiArgs{}; ep.bn_g = bn1_g; ep.bn_b = bn1_b; ep.bn_m = bn1_m; ep.bn_v = bn1_v; ep.out_bf16 = y1;
    gemm_kernel<2, false><<<dim3(C2DIM / 128, MROWS / 128), 256, 0, stream>>>(
        ln_bf, w1_t, MROWS, C2DIM, CDIM, ep);
    ep = EpiArgs{}; ep.bn_g = bn2_g; ep.bn_b = bn2_b; ep.bn_m = bn2_m; ep.bn_v = bn2_v;
    ep.resid = x1; ep.gamma = gamma2; ep.out_f32 = x2;
    gemm_kernel<3, false><<<dim3(CDIM / 128, MROWS / 128), 256, 0, stream>>>(
        y1, w2_t, MROWS, CDIM, C2DIM, ep);

    // --- projection block (3x3 convs, implicit GEMM) ---
    ln_bf16_kernel<<<MROWS, 128, 0, stream>>>(x2, ln3_g, ln3_b, ln_bf);
    ep = EpiArgs{}; ep.bn_g = pbn_g; ep.bn_b = pbn_b; ep.bn_m = pbn_m; ep.bn_v = pbn_v; ep.out_bf16 = p1;
    gemm_kernel<2, true><<<dim3(CDIM / 128, MROWS / 128), 256, 0, stream>>>(
        ln_bf, wc1_t, MROWS, CDIM, 9 * CDIM, ep);
    ep = EpiArgs{}; ep.resid = x2; ep.gamma = gamma3; ep.out_f32 = (float*)d_out;
    gemm_kernel<1, true><<<dim3(CDIM / 128, MROWS / 128), 256, 0, stream>>>(
        p1, wc2_t, MROWS, CDIM, 9 * CDIM, ep);

    tail_kernel<<<1, 32, 0, stream>>>((float*)d_out, Hp, Wp, out_size);
}
